// AdaGML_39221641347166
// MI455X (gfx1250) — matrix operand order repaired and checked by reading, not compiled, not run on hardware
//
#include <hip/hip_runtime.h>
#include <math.h>

// ---------------------------------------------------------------------------
// LightGlue-style matcher for MI455X (gfx1250, wave32, WMMA).
// GEMMs: v_wmma_f32_16x16x32_bf16, 128x64 block tiles, double-buffered LDS
// staged with CDNA5 async global->LDS copies (global_load_async_to_lds_b128,
// tracked by ASYNCcnt per CDNA5 ISA).
// ---------------------------------------------------------------------------

typedef __attribute__((ext_vector_type(16))) __bf16 v16bf;
typedef __attribute__((ext_vector_type(8)))  float  v8f;

#define NKPT   2048
#define NHEAD  4
#define NLAY   9

// 32-bit LDS byte offset of a generic pointer that points into LDS
__device__ __forceinline__ unsigned lds_addr32(const void* p) {
  return (unsigned)(unsigned long long)(uintptr_t)
      (const __attribute__((address_space(3))) char*)p;
}

// ======================= WMMA bf16 GEMM =====================================
// C[M,N] = alpha * A[M,K] x B[K,N] (+ bias[N]) (+= if accum)
// A: bf16 row-major [M,K] (lda); Bt: bf16 row-major [N,K] (ldb) i.e. B^T.
// M multiple of 128, N multiple of 64, K multiple of 32, lda/ldb mult of 8.
// 128 threads = 4 waves; block tile 128x64; wave tile 64x32 = 4x2 WMMA frags.
// Double-buffered LDS staged via async global->LDS b128 copies.

__device__ __forceinline__ int kmapA(int v, int g) {
  // 16-bit A 16x32 fragment: lane group g holds K in {0..7,16..23}+8g
  return (v < 4 ? 2 * v : 16 + 2 * (v - 4)) + 8 * g;
}

__global__ __launch_bounds__(128) void k_gemm_bf16(
    int M, int N, int K,
    const __bf16* __restrict__ A, int lda,
    const __bf16* __restrict__ Bt, int ldb,
    const float* __restrict__ bias,
    float* __restrict__ C, int ldc,
    float alpha, int accum)
{
  (void)M; (void)N;
  __shared__ uint4 As4[2][512];   // 128 rows x 32 bf16 per buffer
  __shared__ uint4 Bs4[2][256];   // 64  rows x 32 bf16 per buffer

  int tid  = threadIdx.x;
  int wave = tid >> 5;
  int lane = tid & 31;
  int g    = lane >> 4;     // lane group (0/1)
  int r    = lane & 15;
  int wm   = (wave >> 1) * 64;
  int wn   = (wave & 1) * 32;
  size_t tile_m = (size_t)blockIdx.y * 128;
  size_t tile_n = (size_t)blockIdx.x * 64;

  v8f acc[4][2] = {};

  // async-stage one K-chunk (32 cols) of A (128 rows) and Bt (64 rows)
  auto stage = [&](int buf, int k0) {
    for (int c = tid; c < 512; c += 128) {
      unsigned l = lds_addr32(&As4[buf][c]);
      const __bf16* gp = A + (tile_m + (c >> 2)) * (size_t)lda + k0 + (c & 3) * 8;
      asm volatile("global_load_async_to_lds_b128 %0, %1, off"
                   :: "v"(l), "v"((unsigned long long)(uintptr_t)gp) : "memory");
    }
    for (int c = tid; c < 256; c += 128) {
      unsigned l = lds_addr32(&Bs4[buf][c]);
      const __bf16* gp = Bt + (tile_n + (c >> 2)) * (size_t)ldb + k0 + (c & 3) * 8;
      asm volatile("global_load_async_to_lds_b128 %0, %1, off"
                   :: "v"(l), "v"((unsigned long long)(uintptr_t)gp) : "memory");
    }
  };

  stage(0, 0);
  int nk = K >> 5;
  for (int t = 0; t < nk; ++t) {
    // my async writes into buf(t&1) must land before anyone reads them
    asm volatile("s_wait_asynccnt 0x0" ::: "memory");
    __syncthreads();   // also orders prev-iter fragment reads (dscnt) vs new writes
    int cur = t & 1;
    if (t + 1 < nk) stage(cur ^ 1, (t + 1) << 5);

    const unsigned* As32 = reinterpret_cast<const unsigned*>(As4[cur]);
    const unsigned* Bs32 = reinterpret_cast<const unsigned*>(Bs4[cur]);
    union Frag { unsigned u[8]; v16bf v; };
    Frag a[4], b[2];
    for (int i = 0; i < 4; ++i) {
      int row = wm + 16 * i + r;
      for (int v = 0; v < 8; ++v)
        a[i].u[v] = As32[row * 16 + (kmapA(v, g) >> 1)];
    }
    for (int j = 0; j < 2; ++j) {
      int row = wn + 16 * j + r;
      for (int v = 0; v < 8; ++v)
        b[j].u[v] = Bs32[row * 16 + (kmapA(v, g) >> 1)];   // B lane group g: K in {0..7,16..23}+8g
    }
    for (int i = 0; i < 4; ++i)
      for (int j = 0; j < 2; ++j)
        acc[i][j] = __builtin_amdgcn_wmma_f32_16x16x32_bf16(
            false, a[i].v, false, b[j].v, (short)0, acc[i][j], false, false);
    __syncthreads();
  }

  for (int i = 0; i < 4; ++i)
    for (int j = 0; j < 2; ++j)
      for (int rr = 0; rr < 8; ++rr) {
        size_t mrow = tile_m + wm + 16 * i + 8 * g + rr;   // D layout: M = r + 8*laneGroup
        size_t ncol = tile_n + wn + 16 * j + r;            // N = lane & 15
        float val = acc[i][j][rr] * alpha;
        if (bias) val += bias[ncol];
        size_t o = mrow * (size_t)ldc + ncol;
        if (accum) C[o] += val; else C[o] = val;
      }
}

// ======================= elementwise / reduction kernels ====================

__global__ __launch_bounds__(256) void k_f2bf(const float* __restrict__ s,
                                              __bf16* __restrict__ d, int n) {
  int i = blockIdx.x * 256 + threadIdx.x;
  if (i < n) d[i] = (__bf16)s[i];
}

// W [K,N] f32 -> Wt [N,K] bf16
__global__ __launch_bounds__(256) void k_wtrans(const float* __restrict__ W,
                                                __bf16* __restrict__ Wt, int K, int N) {
  int i = blockIdx.x * 256 + threadIdx.x;
  if (i >= K * N) return;
  int n = i / K, k = i - n * K;
  Wt[i] = (__bf16)W[(size_t)k * N + n];
}

// Learnable Fourier pos-enc: kpts[N,2] @ Wr[2,32] -> cos/sin repeated x2 -> [N,64]
__global__ __launch_bounds__(256) void k_pos_enc(const float* __restrict__ kpts,
                                                 const float* __restrict__ Wr,
                                                 float* __restrict__ cosb,
                                                 float* __restrict__ sinb) {
  int i = blockIdx.x * 256 + threadIdx.x;   // NKPT*32 threads
  int n = i >> 5, c = i & 31;
  float p = kpts[n * 2] * Wr[c] + kpts[n * 2 + 1] * Wr[32 + c];
  float cv = cosf(p), sv = sinf(p);
  cosb[n * 64 + 2 * c] = cv; cosb[n * 64 + 2 * c + 1] = cv;
  sinb[n * 64 + 2 * c] = sv; sinb[n * 64 + 2 * c + 1] = sv;
}

// qkv[N,768] (layout h*192 + d*3 + s) -> roped Q,K bf16 [H][N][64], V^T bf16 [H][64][N]
__global__ __launch_bounds__(256) void k_self_split(const float* __restrict__ qkv,
                                                    const float* __restrict__ cosb,
                                                    const float* __restrict__ sinb,
                                                    __bf16* __restrict__ Qh,
                                                    __bf16* __restrict__ Kh,
                                                    __bf16* __restrict__ Vt) {
  int idx = blockIdx.x * 256 + threadIdx.x;        // NHEAD*NKPT*64
  int d = idx & 63;
  int n = (idx >> 6) & (NKPT - 1);
  int h = idx >> 17;
  size_t base = (size_t)n * 768 + h * 192;
  float q  = qkv[base + d * 3 + 0];
  float k  = qkv[base + d * 3 + 1];
  float v  = qkv[base + d * 3 + 2];
  int dp = d ^ 1;
  float qp = qkv[base + dp * 3 + 0];
  float kp = qkv[base + dp * 3 + 1];
  float rq = (d & 1) ? qp : -qp;   // rotate_half: even d -> -x[d+1], odd d -> x[d-1]
  float rk = (d & 1) ? kp : -kp;
  float c = cosb[n * 64 + d], s = sinb[n * 64 + d];
  size_t hb = (size_t)h * NKPT * 64;
  Qh[hb + (size_t)n * 64 + d] = (__bf16)(q * c + rq * s);
  Kh[hb + (size_t)n * 64 + d] = (__bf16)(k * c + rk * s);
  Vt[(size_t)h * 64 * NKPT + (size_t)d * NKPT + n] = (__bf16)v;
}

// src[N,256] -> bf16 heads [H][N][64] (scaled) or transposed [H][64][N]
__global__ __launch_bounds__(256) void k_cross_split(const float* __restrict__ src,
                                                     float scale,
                                                     __bf16* __restrict__ dst,
                                                     int transposed) {
  int idx = blockIdx.x * 256 + threadIdx.x;   // NKPT*256
  int n = idx >> 8, c = idx & 255;
  int h = c >> 6, d = c & 63;
  float v = src[idx] * scale;
  if (transposed) dst[(size_t)h * 64 * NKPT + (size_t)d * NKPT + n] = (__bf16)v;
  else            dst[(size_t)h * NKPT * 64 + (size_t)n * 64 + d]   = (__bf16)v;
}

__global__ __launch_bounds__(256) void k_concat(const float* __restrict__ x,
                                                const float* __restrict__ m,
                                                __bf16* __restrict__ cat) {
  int idx = blockIdx.x * 256 + threadIdx.x;   // NKPT*512
  int n = idx >> 9, c = idx & 511;
  float v = (c < 256) ? x[(size_t)n * 256 + c] : m[(size_t)n * 256 + (c - 256)];
  cat[idx] = (__bf16)v;
}

// row softmax of S[rows, cols] f32 -> P bf16
__global__ __launch_bounds__(256) void k_softmax_row(const float* __restrict__ S,
                                                     __bf16* __restrict__ P, int cols) {
  int row = blockIdx.x, tid = threadIdx.x;
  const float* s = S + (size_t)row * cols;
  __bf16* p = P + (size_t)row * cols;
  __shared__ float red[256];
  float mx = -3.0e38f;
  for (int j = tid; j < cols; j += 256) mx = fmaxf(mx, s[j]);
  red[tid] = mx; __syncthreads();
  for (int st = 128; st > 0; st >>= 1) { if (tid < st) red[tid] = fmaxf(red[tid], red[tid + st]); __syncthreads(); }
  mx = red[0]; __syncthreads();
  float sum = 0.f;
  for (int j = tid; j < cols; j += 256) sum += expf(s[j] - mx);
  red[tid] = sum; __syncthreads();
  for (int st = 128; st > 0; st >>= 1) { if (tid < st) red[tid] += red[tid + st]; __syncthreads(); }
  float inv = 1.f / red[0];
  for (int j = tid; j < cols; j += 256) p[j] = (__bf16)(expf(s[j] - mx) * inv);
}

// fused LayerNorm(512) + exact GELU, f32 in -> bf16 out
__global__ __launch_bounds__(256) void k_ln_gelu(const float* __restrict__ H,
                                                 const float* __restrict__ gg,
                                                 const float* __restrict__ bb,
                                                 __bf16* __restrict__ out) {
  int row = blockIdx.x, tid = threadIdx.x;
  const float* h = H + (size_t)row * 512;
  __shared__ float red[256];
  float x0 = h[tid], x1 = h[tid + 256];
  red[tid] = x0 + x1; __syncthreads();
  for (int st = 128; st > 0; st >>= 1) { if (tid < st) red[tid] += red[tid + st]; __syncthreads(); }
  float mean = red[0] * (1.f / 512.f);
  __syncthreads();
  float d0 = x0 - mean, d1 = x1 - mean;
  red[tid] = d0 * d0 + d1 * d1; __syncthreads();
  for (int st = 128; st > 0; st >>= 1) { if (tid < st) red[tid] += red[tid + st]; __syncthreads(); }
  float inv = rsqrtf(red[0] * (1.f / 512.f) + 1e-5f);
  float y0 = d0 * inv * gg[tid] + bb[tid];
  float y1 = d1 * inv * gg[tid + 256] + bb[tid + 256];
  out[(size_t)row * 512 + tid]       = (__bf16)(0.5f * y0 * (1.f + erff(y0 * 0.7071067811865475f)));
  out[(size_t)row * 512 + tid + 256] = (__bf16)(0.5f * y1 * (1.f + erff(y1 * 0.7071067811865475f)));
}

// dual-softmax stats over the virtual 2049x2049 matrix M (last row/col = binv)
__global__ __launch_bounds__(256) void k_ds_stats(const float* __restrict__ S,
                                                  const float* __restrict__ binp,
                                                  float* __restrict__ omax,
                                                  float* __restrict__ olse,
                                                  int transposed) {
  int i = blockIdx.x, tid = threadIdx.x;
  float binv = *binp;
  __shared__ float red[256];
  float mx = -3.0e38f;
  for (int j = tid; j < NKPT + 1; j += 256) {
    float v = (i < NKPT && j < NKPT)
        ? (transposed ? S[(size_t)j * NKPT + i] : S[(size_t)i * NKPT + j]) : binv;
    mx = fmaxf(mx, v);
  }
  red[tid] = mx; __syncthreads();
  for (int st = 128; st > 0; st >>= 1) { if (tid < st) red[tid] = fmaxf(red[tid], red[tid + st]); __syncthreads(); }
  mx = red[0]; __syncthreads();
  float sum = 0.f;
  for (int j = tid; j < NKPT + 1; j += 256) {
    float v = (i < NKPT && j < NKPT)
        ? (transposed ? S[(size_t)j * NKPT + i] : S[(size_t)i * NKPT + j]) : binv;
    sum += expf(v - mx);
  }
  red[tid] = sum; __syncthreads();
  for (int st = 128; st > 0; st >>= 1) { if (tid < st) red[tid] += red[tid + st]; __syncthreads(); }
  if (tid == 0) { omax[i] = mx; olse[i] = logf(red[0]); }
}

__global__ __launch_bounds__(256) void k_ds_out(const float* __restrict__ S,
                                                const float* __restrict__ binp,
                                                const float* __restrict__ rmax,
                                                const float* __restrict__ rlse,
                                                const float* __restrict__ cmax,
                                                const float* __restrict__ clse,
                                                float* __restrict__ out) {
  size_t idx = (size_t)blockIdx.x * 256 + threadIdx.x;
  const size_t tot = (size_t)(NKPT + 1) * (NKPT + 1);
  if (idx >= tot) return;
  int i = (int)(idx / (NKPT + 1));
  int j = (int)(idx - (size_t)i * (NKPT + 1));
  float binv = *binp;
  float v = (i < NKPT && j < NKPT) ? S[(size_t)i * NKPT + j] : binv;
  out[idx] = expf(2.f * v - rmax[i] - rlse[i] - cmax[j] - clse[j]);
}

// ======================= host orchestration =================================

extern "C" void kernel_launch(void* const* d_in, const int* in_sizes, int n_in,
                              void* d_out, int out_size, void* d_ws, size_t ws_size,
                              hipStream_t stream) {
  (void)in_sizes; (void)n_in; (void)out_size; (void)ws_size;

  // ---- parameter flattening (jax pytree: dict keys sorted alphabetically) ----
  // 0..3: desc0, desc1, kpts0, kpts1 ; 4: Wr ; 5: bin
  // 6 + 12*l (cross l): lin1.b, lin1.w, lin2.b, lin2.w, ln.b, ln.g,
  //                     proj.b, proj.w, qk.b, qk.w, v.b, v.w
  // 114: input_proj.b  115: input_proj.w  116: out.b  117: out.w
  // 118 + 10*l (self l): lin1.b, lin1.w, lin2.b, lin2.w, ln.b, ln.g,
  //                      proj.b, proj.w, qkv.b, qkv.w
  auto F = [&](int i) { return (const float*)d_in[i]; };
  const float* desc0 = F(0); const float* desc1 = F(1);
  const float* kpts0 = F(2); const float* kpts1 = F(3);
  const float* Wr = F(4);    const float* binp = F(5);
  const int CB = 6, SB = 118;
  const float* ipb = F(114); const float* ipw = F(115);
  const float* outb = F(116); const float* outw = F(117);

  // ---- workspace carve-out (256B aligned) ----
  char* wsb = (char*)d_ws; size_t off = 0;
  auto alloc = [&](size_t bytes) -> char* {
    char* p = wsb + off; off += (bytes + 255) & ~(size_t)255; return p;
  };
  __bf16* wt_input = (__bf16*)alloc(256 * 128 * 2);
  __bf16 *wt_qkv[NLAY], *wt_sproj[NLAY], *wt_sl1[NLAY], *wt_sl2[NLAY];
  __bf16 *wt_qk[NLAY], *wt_v[NLAY], *wt_cproj[NLAY], *wt_cl1[NLAY], *wt_cl2[NLAY];
  for (int l = 0; l < NLAY; ++l) {
    wt_qkv[l]   = (__bf16*)alloc(768 * 256 * 2);
    wt_sproj[l] = (__bf16*)alloc(256 * 256 * 2);
    wt_sl1[l]   = (__bf16*)alloc(512 * 512 * 2);
    wt_sl2[l]   = (__bf16*)alloc(256 * 512 * 2);
    wt_qk[l]    = (__bf16*)alloc(256 * 256 * 2);
    wt_v[l]     = (__bf16*)alloc(256 * 256 * 2);
    wt_cproj[l] = (__bf16*)alloc(256 * 256 * 2);
    wt_cl1[l]   = (__bf16*)alloc(512 * 512 * 2);
    wt_cl2[l]   = (__bf16*)alloc(256 * 512 * 2);
  }
  __bf16* wt_out = (__bf16*)alloc(256 * 256 * 2);
  float* cos0 = (float*)alloc((size_t)NKPT * 64 * 4);
  float* sin0 = (float*)alloc((size_t)NKPT * 64 * 4);
  float* cos1 = (float*)alloc((size_t)NKPT * 64 * 4);
  float* sin1 = (float*)alloc((size_t)NKPT * 64 * 4);
  float* x0   = (float*)alloc((size_t)NKPT * 256 * 4);
  float* x1   = (float*)alloc((size_t)NKPT * 256 * 4);
  float* qkvf = (float*)alloc((size_t)NKPT * 768 * 4);
  float* omsg = (float*)alloc((size_t)NKPT * 256 * 4);
  float* mbuf = (float*)alloc((size_t)NKPT * 256 * 4);
  float* h1b  = (float*)alloc((size_t)NKPT * 512 * 4);
  float* Sb   = (float*)alloc((size_t)NKPT * NKPT * 4);
  float* md0f = (float*)alloc((size_t)NKPT * 256 * 4);
  float* md1f = (float*)alloc((size_t)NKPT * 256 * 4);
  float* rmax = (float*)alloc(2052 * 4); float* rlse = (float*)alloc(2052 * 4);
  float* cmax = (float*)alloc(2052 * 4); float* clse = (float*)alloc(2052 * 4);
  __bf16* abuf = (__bf16*)alloc((size_t)NKPT * 512 * 2);
  __bf16* cbuf = (__bf16*)alloc((size_t)NKPT * 512 * 2);
  __bf16* Qh   = (__bf16*)alloc((size_t)NHEAD * NKPT * 64 * 2);
  __bf16* Kh   = (__bf16*)alloc((size_t)NHEAD * NKPT * 64 * 2);
  __bf16* Vt0  = (__bf16*)alloc((size_t)NHEAD * 64 * NKPT * 2);
  __bf16* Vt1  = (__bf16*)alloc((size_t)NHEAD * 64 * NKPT * 2);
  __bf16* Pb   = (__bf16*)alloc((size_t)NKPT * NKPT * 2);
  __bf16* md0b = (__bf16*)alloc((size_t)NKPT * 256 * 2);
  __bf16* md1b = (__bf16*)alloc((size_t)NKPT * 256 * 2);

  auto gemm = [&](int M, int N, int K, const __bf16* A, int lda,
                  const __bf16* Bt, int ldb, const float* bias,
                  float* C, int ldc, float alpha, int accum) {
    dim3 grid(N / 64, M / 128);
    k_gemm_bf16<<<grid, dim3(128), 0, stream>>>(M, N, K, A, lda, Bt, ldb, bias, C, ldc, alpha, accum);
  };
  auto f2bf = [&](const float* s, __bf16* d, int n) {
    k_f2bf<<<(n + 255) / 256, 256, 0, stream>>>(s, d, n);
  };
  auto convT = [&](const float* W, __bf16* Wt, int K, int N) {
    int tot = K * N;
    k_wtrans<<<(tot + 255) / 256, 256, 0, stream>>>(W, Wt, K, N);
  };

  // ---- convert all weights (per call; deterministic) ----
  convT(ipw, wt_input, 128, 256);
  convT(outw, wt_out, 256, 256);
  for (int l = 0; l < NLAY; ++l) {
    int s = SB + 10 * l, c = CB + 12 * l;
    convT(F(s + 9), wt_qkv[l], 256, 768);
    convT(F(s + 7), wt_sproj[l], 256, 256);
    convT(F(s + 1), wt_sl1[l], 512, 512);
    convT(F(s + 3), wt_sl2[l], 512, 256);
    convT(F(c + 9), wt_qk[l], 256, 256);
    convT(F(c + 11), wt_v[l], 256, 256);
    convT(F(c + 7), wt_cproj[l], 256, 256);
    convT(F(c + 1), wt_cl1[l], 512, 512);
    convT(F(c + 3), wt_cl2[l], 512, 256);
  }

  // ---- positional encodings ----
  k_pos_enc<<<(NKPT * 32) / 256, 256, 0, stream>>>(kpts0, Wr, cos0, sin0);
  k_pos_enc<<<(NKPT * 32) / 256, 256, 0, stream>>>(kpts1, Wr, cos1, sin1);

  // ---- input projection ----
  f2bf(desc0, abuf, NKPT * 128);
  gemm(NKPT, 256, 128, abuf, 128, wt_input, 128, ipb, x0, 256, 1.f, 0);
  f2bf(desc1, abuf, NKPT * 128);
  gemm(NKPT, 256, 128, abuf, 128, wt_input, 128, ipb, x1, 256, 1.f, 0);

  auto attention = [&](const __bf16* Q, const __bf16* Kk, const __bf16* Vt, float alpha) {
    for (int h = 0; h < NHEAD; ++h) {
      gemm(NKPT, NKPT, 64, Q + (size_t)h * NKPT * 64, 64,
           Kk + (size_t)h * NKPT * 64, 64, nullptr, Sb, NKPT, alpha, 0);
      k_softmax_row<<<NKPT, 256, 0, stream>>>(Sb, Pb, NKPT);
      gemm(NKPT, 64, NKPT, Pb, NKPT, Vt + (size_t)h * 64 * NKPT, NKPT,
           nullptr, omsg + h * 64, 256, 1.f, 0);
    }
  };
  auto proj_mlp = [&](float* x, const __bf16* wproj, const float* pb,
                      const __bf16* w1, const float* b1,
                      const float* lng, const float* lnb,
                      const __bf16* w2, const float* b2) {
    f2bf(omsg, abuf, NKPT * 256);
    gemm(NKPT, 256, 256, abuf, 256, wproj, 256, pb, mbuf, 256, 1.f, 0);
    k_concat<<<(NKPT * 512) / 256, 256, 0, stream>>>(x, mbuf, cbuf);
    gemm(NKPT, 512, 512, cbuf, 512, w1, 512, b1, h1b, 512, 1.f, 0);
    k_ln_gelu<<<NKPT, 256, 0, stream>>>(h1b, lng, lnb, abuf);
    gemm(NKPT, 256, 512, abuf, 512, w2, 512, b2, x, 256, 1.f, 1);  // residual accumulate
  };

  const float cs = 0.3535533905932738f;  // DH^-0.25

  for (int l = 0; l < NLAY; ++l) {
    int s = SB + 10 * l, c = CB + 12 * l;
    // ---------- self attention (both images) ----------
    float* xs[2] = {x0, x1};
    const float* cb[2] = {cos0, cos1};
    const float* sn[2] = {sin0, sin1};
    for (int im = 0; im < 2; ++im) {
      f2bf(xs[im], abuf, NKPT * 256);
      gemm(NKPT, 768, 256, abuf, 256, wt_qkv[l], 256, F(s + 8), qkvf, 768, 1.f, 0);
      k_self_split<<<(NHEAD * NKPT * 64) / 256, 256, 0, stream>>>(qkvf, cb[im], sn[im], Qh, Kh, Vt0);
      attention(Qh, Kh, Vt0, 0.125f);  // DH^-0.5
      proj_mlp(xs[im], wt_sproj[l], F(s + 6), wt_sl1[l], F(s + 0),
               F(s + 5), F(s + 4), wt_sl2[l], F(s + 2));
    }
    // ---------- cross attention ----------
    f2bf(x0, abuf, NKPT * 256);
    gemm(NKPT, 256, 256, abuf, 256, wt_qk[l], 256, F(c + 8), qkvf, 256, 1.f, 0);
    k_cross_split<<<(NKPT * 256) / 256, 256, 0, stream>>>(qkvf, cs, Qh, 0);
    gemm(NKPT, 256, 256, abuf, 256, wt_v[l], 256, F(c + 10), qkvf, 256, 1.f, 0);
    k_cross_split<<<(NKPT * 256) / 256, 256, 0, stream>>>(qkvf, 1.f, Vt0, 1);
    f2bf(x1, abuf, NKPT * 256);
    gemm(NKPT, 256, 256, abuf, 256, wt_qk[l], 256, F(c + 8), qkvf, 256, 1.f, 0);
    k_cross_split<<<(NKPT * 256) / 256, 256, 0, stream>>>(qkvf, cs, Kh, 0);
    gemm(NKPT, 256, 256, abuf, 256, wt_v[l], 256, F(c + 10), qkvf, 256, 1.f, 0);
    k_cross_split<<<(NKPT * 256) / 256, 256, 0, stream>>>(qkvf, 1.f, Vt1, 1);
    // direction 0 -> x0 ; direction 1 -> x1 (sim recomputed transposed)
    attention(Qh, Kh, Vt1, 1.f);
    proj_mlp(x0, wt_cproj[l], F(c + 6), wt_cl1[l], F(c + 0),
             F(c + 5), F(c + 4), wt_cl2[l], F(c + 2));
    attention(Kh, Qh, Vt0, 1.f);
    proj_mlp(x1, wt_cproj[l], F(c + 6), wt_cl1[l], F(c + 0),
             F(c + 5), F(c + 4), wt_cl2[l], F(c + 2));
  }

  // ---- final descriptors, distance matrix, dual softmax ----
  f2bf(x0, abuf, NKPT * 256);
  gemm(NKPT, 256, 256, abuf, 256, wt_out, 256, outb, md0f, 256, 1.f, 0);
  f2bf(x1, abuf, NKPT * 256);
  gemm(NKPT, 256, 256, abuf, 256, wt_out, 256, outb, md1f, 256, 1.f, 0);
  f2bf(md0f, md0b, NKPT * 256);
  f2bf(md1f, md1b, NKPT * 256);
  gemm(NKPT, NKPT, 256, md0b, 256, md1b, 256, nullptr, Sb, NKPT, 0.0625f, 0);  // /sqrt(256)

  k_ds_stats<<<NKPT + 1, 256, 0, stream>>>(Sb, binp, rmax, rlse, 0);
  k_ds_stats<<<NKPT + 1, 256, 0, stream>>>(Sb, binp, cmax, clse, 1);
  size_t tot = (size_t)(NKPT + 1) * (NKPT + 1);
  k_ds_out<<<(unsigned)((tot + 255) / 256), 256, 0, stream>>>(
      Sb, binp, rmax, rlse, cmax, clse, (float*)d_out);
}
